// MultiheadISAAttention_52802327937246
// MI455X (gfx1250) — matrix operand order repaired and checked by reading, not compiled, not run on hardware
//
#include <hip/hip_runtime.h>

typedef __attribute__((ext_vector_type(16))) __bf16 bf16x16;
typedef __attribute__((ext_vector_type(8)))  __bf16 bf16x8;
typedef __attribute__((ext_vector_type(8)))  float  f32x8;

namespace cfg {
constexpr int BATCH = 8;
constexpr int IMG   = 126;
constexpr int C     = 512;
constexpr int WSZ   = 7;
constexpr int NHEAD = 8;
constexpr int HD    = 64;       // C / NHEAD
constexpr int QDIM  = IMG / WSZ;            // 18
constexpr int NW    = QDIM * QDIM;          // 324 windows per image
constexpr int L     = WSZ * WSZ;            // 49
constexpr int BW    = BATCH * NW;           // 2592 total windows
constexpr int MTOT  = L * BW;               // 127008 rows
constexpr int C3    = 3 * C;                // 1536
}

// ---------------------------------------------------------------------------
// CDNA5 async global->LDS copy (16B per lane), tracked by ASYNCcnt.
// ---------------------------------------------------------------------------
__device__ __forceinline__ void async_g2l_b128(unsigned lds_off, const void* gaddr) {
  asm volatile("global_load_async_to_lds_b128 %0, %1, off"
               :: "v"(lds_off), "v"(gaddr)
               : "memory");
}
__device__ __forceinline__ void wait_async0() {
  asm volatile("s_wait_asynccnt 0" ::: "memory");
}
__device__ __forceinline__ unsigned lds_off(const void* p) {
  return (unsigned)(size_t)p;   // LDS aperture: addr[31:0] is the LDS offset
}

// ---------------------------------------------------------------------------
// WMMA fragment helpers (CDNA5 wave32 16x16x32 bf16 layouts, ISA 7.12.2)
// ---------------------------------------------------------------------------
__device__ __forceinline__ bf16x16 frag_a(const __bf16* mat, int ld, int row0, int kc) {
  const int lane = threadIdx.x & 31;
  const int l16 = lane & 15, hf = lane >> 4;
  const __bf16* p = mat + (size_t)(row0 + l16) * ld + kc + hf * 8;
  bf16x8 lo = *(const bf16x8*)p;
  bf16x8 hi = *(const bf16x8*)(p + 16);
  bf16x16 r;
#pragma unroll
  for (int i = 0; i < 8; ++i) { r[i] = lo[i]; r[i + 8] = hi[i]; }
  return r;
}

__device__ __forceinline__ bf16x16 frag_b(const __bf16* bt, int ld, int col0, int kc) {
  const int lane = threadIdx.x & 31;
  const int l16 = lane & 15, hf = lane >> 4;
  const __bf16* p = bt + (size_t)(col0 + l16) * ld + kc + hf * 8;
  bf16x8 lo = *(const bf16x8*)p;
  bf16x8 hi = *(const bf16x8*)(p + 16);
  bf16x16 r;
#pragma unroll
  for (int i = 0; i < 8; ++i) { r[i] = lo[i]; r[i + 8] = hi[i]; }
  return r;
}

__device__ __forceinline__ f32x8 wmma_bf16(bf16x16 a, bf16x16 b, f32x8 c) {
  return __builtin_amdgcn_wmma_f32_16x16x32_bf16(false, a, false, b, (short)0, c,
                                                 false, false);
}

union bf4pack { uint2 u; __bf16 b[4]; };

// ---------------------------------------------------------------------------
// Kernel 0: one-time fp32 -> bf16 conversion of both weight matrices.
// ---------------------------------------------------------------------------
__global__ __launch_bounds__(256) void convert_weights_kernel(
    const float* __restrict__ w_in, const float* __restrict__ w_out,
    __bf16* __restrict__ wbf_in, __bf16* __restrict__ wbf_out) {
  using namespace cfg;
  constexpr int G1 = C3 * C / 4;   // 196608 groups of 4
  constexpr int G2 = C * C / 4;    // 65536 groups of 4
  int g = blockIdx.x * 256 + threadIdx.x;
  const float* src; __bf16* dst; int idx;
  if (g < G1) { src = w_in; dst = wbf_in; idx = g; }
  else if (g < G1 + G2) { src = w_out; dst = wbf_out; idx = g - G1; }
  else return;
  float4 v = ((const float4*)src)[idx];
  bf4pack pk;
  pk.b[0] = (__bf16)v.x; pk.b[1] = (__bf16)v.y;
  pk.b[2] = (__bf16)v.z; pk.b[3] = (__bf16)v.w;
  ((uint2*)dst)[idx] = pk.u;
}

// ---------------------------------------------------------------------------
// Kernel 1: qkv = window_permute(x) @ W_in^T + b_in   (q scaled by HD^-0.5)
// Block: 256 threads (8 waves).  Block tile: 32 rows x 256 cols per pass
// (6 passes over 1536).  Wave tile 32x32 -> 4 WMMA per K-step.
// x panel staged once (fp32->bf16); weights async global->LDS (bf16).
// Output layout: qkv[w][p][h][l][d]  (bf16)
// ---------------------------------------------------------------------------
__global__ __launch_bounds__(256) void qkv_proj_kernel(
    const float* __restrict__ x, const __bf16* __restrict__ wbf,
    const float* __restrict__ b_in, __bf16* __restrict__ qkv) {
  using namespace cfg;
  __shared__ __bf16 As[32 * 520];   // 32 x 512 panel (stride 1040B, 16B aligned)
  __shared__ __bf16 Bs[256 * 40];   // 256 cols x 32 K (stride 80B, 16B aligned)

  const int tid = threadIdx.x;
  const int lane = tid & 31, l16 = lane & 15, hf = lane >> 4;
  const int wv = tid >> 5;               // 0..7 -> 32-col stripe
  const int m0 = blockIdx.x * 32;

  // --- stage A panel: window-permuted row -> x source row (convert to bf16)
  {
    const int arow = tid >> 3;          // 0..31
    const int acol0 = (tid & 7) * 4;    // 0..28
    int r = m0 + arow;
    int l = r / BW, w = r % BW;
    int b = w / NW, rem = w % NW;
    int qy = rem / QDIM, qx = rem % QDIM;
    int ph = l / WSZ, pw = l % WSZ;
    const float* asrc =
        x + ((size_t)b * (IMG * IMG) + (size_t)((qy * WSZ + ph) * IMG + qx * WSZ + pw)) * C;
#pragma unroll
    for (int j = 0; j < 16; ++j) {
      int col = acol0 + j * 32;
      float4 v = *(const float4*)(asrc + col);
      bf4pack pk;
      pk.b[0] = (__bf16)v.x; pk.b[1] = (__bf16)v.y;
      pk.b[2] = (__bf16)v.z; pk.b[3] = (__bf16)v.w;
      *(uint2*)(As + (size_t)arow * 520 + col) = pk.u;
    }
  }
  __syncthreads();

  const unsigned bsOff = lds_off(Bs) + tid * 80;   // this thread's B row (80B)

  for (int nb = 0; nb < C3 / 256; ++nb) {          // 6 N tiles of 256
    f32x8 acc[2][2] = {};
    const __bf16* bsrc = wbf + (size_t)(nb * 256 + tid) * C;
    for (int kt = 0; kt < C / 32; ++kt) {          // 16 K steps of 32
      // async-stage B tile: one 64B row slice per thread
#pragma unroll
      for (int j = 0; j < 4; ++j)
        async_g2l_b128(bsOff + j * 16, bsrc + kt * 32 + j * 8);
      wait_async0();
      __syncthreads();
      bf16x16 a0 = frag_a(As, 520, 0, kt * 32);
      bf16x16 a1 = frag_a(As, 520, 16, kt * 32);
      bf16x16 b0 = frag_b(Bs, 40, wv * 32, 0);
      bf16x16 b1 = frag_b(Bs, 40, wv * 32 + 16, 0);
      acc[0][0] = wmma_bf16(a0, b0, acc[0][0]);
      acc[0][1] = wmma_bf16(a0, b1, acc[0][1]);
      acc[1][0] = wmma_bf16(a1, b0, acc[1][0]);
      acc[1][1] = wmma_bf16(a1, b1, acc[1][1]);
      __syncthreads();
    }
    // epilogue: bias, q-scale, scatter to [w][p][h][l][d]
#pragma unroll
    for (int mi = 0; mi < 2; ++mi) {
#pragma unroll
      for (int ci = 0; ci < 2; ++ci) {
#pragma unroll
        for (int i = 0; i < 8; ++i) {
          int rr = m0 + mi * 16 + hf * 8 + i;
          int cg = nb * 256 + wv * 32 + ci * 16 + l16;
          float v = acc[mi][ci][i] + b_in[cg];
          int p = cg >> 9;                  // 0=q,1=k,2=v
          if (p == 0) v *= 0.125f;          // HD^-0.5, HD=64
          int hh = (cg >> 6) & 7, d = cg & 63;
          int ll = rr / BW, ww = rr % BW;
          qkv[((((size_t)ww * 3 + p) * NHEAD + hh) * L + ll) * HD + d] = (__bf16)v;
        }
      }
    }
  }
}

// ---------------------------------------------------------------------------
// Kernel 2: per (window, head) attention.  Block = 128 threads (4 waves).
// Q/K async global->LDS; V transposed via scalar stores; S = QK^T (+rpe),
// softmax, O = P V.
// ---------------------------------------------------------------------------
__global__ __launch_bounds__(128) void attn_kernel(
    const __bf16* __restrict__ qkv, const float* __restrict__ rpe_table,
    const int* __restrict__ rel_index, __bf16* __restrict__ ao) {
  using namespace cfg;
  __shared__ __bf16 Qs[64 * 72];
  __shared__ __bf16 Ks[64 * 72];
  __shared__ __bf16 Vt[64 * 72];    // transposed: Vt[d][l]
  __shared__ __bf16 Ps[64 * 72];    // softmax probs (bf16)
  __shared__ float  Ss[64 * 68];    // scores f32
  __shared__ float  Rp[L * L];      // rpe bias for this head

  const int tid = threadIdx.x;
  const int lane = tid & 31, l16 = lane & 15, hf = lane >> 4;
  const int wv = tid >> 5;
  const int w = blockIdx.x >> 3;    // window
  const int h = blockIdx.x & 7;     // head

  const __bf16* qg = qkv + (((size_t)w * 3 + 0) * NHEAD + h) * (L * HD);
  const __bf16* kg = qkv + (((size_t)w * 3 + 1) * NHEAD + h) * (L * HD);
  const __bf16* vg = qkv + (((size_t)w * 3 + 2) * NHEAD + h) * (L * HD);

  // async-stage Q, K valid region (49 rows x 64 = 392 x 16B chunks each)
  const unsigned qOff = lds_off(Qs), kOff = lds_off(Ks);
  for (int g = tid; g < L * 8; g += 128) {
    int l = g >> 3, c8 = (g & 7) * 8;
    unsigned o = (unsigned)(l * 72 + c8) * 2;
    async_g2l_b128(qOff + o, qg + (size_t)l * HD + c8);
    async_g2l_b128(kOff + o, kg + (size_t)l * HD + c8);
  }
  // V transpose + zero padding (rows 49..63 of Q/K, cols 49..63 of Vt)
  for (int g = tid; g < 64 * 16; g += 128) {
    int l = g >> 4, d4 = (g & 15) * 4;
    if (l < L) {
      const __bf16* vp = vg + (size_t)l * HD + d4;
      Vt[(size_t)(d4 + 0) * 72 + l] = vp[0];
      Vt[(size_t)(d4 + 1) * 72 + l] = vp[1];
      Vt[(size_t)(d4 + 2) * 72 + l] = vp[2];
      Vt[(size_t)(d4 + 3) * 72 + l] = vp[3];
    } else {
      uint2 z = make_uint2(0u, 0u);
      *(uint2*)(Qs + (size_t)l * 72 + d4) = z;
      *(uint2*)(Ks + (size_t)l * 72 + d4) = z;
      __bf16 zb = (__bf16)0.0f;
      Vt[(size_t)(d4 + 0) * 72 + l] = zb;
      Vt[(size_t)(d4 + 1) * 72 + l] = zb;
      Vt[(size_t)(d4 + 2) * 72 + l] = zb;
      Vt[(size_t)(d4 + 3) * 72 + l] = zb;
    }
  }
  for (int idx = tid; idx < L * L; idx += 128)
    Rp[idx] = rpe_table[rel_index[idx] * NHEAD + h];
  wait_async0();
  __syncthreads();

  // ---- S = Q K^T + rpe (masked) ----
  const int row0 = wv * 16;
  bf16x16 a0 = frag_a(Qs, 72, row0, 0);
  bf16x16 a1 = frag_a(Qs, 72, row0, 32);
#pragma unroll
  for (int ct = 0; ct < 4; ++ct) {
    f32x8 acc = {};
    acc = wmma_bf16(a0, frag_b(Ks, 72, ct * 16, 0), acc);
    acc = wmma_bf16(a1, frag_b(Ks, 72, ct * 16, 32), acc);
#pragma unroll
    for (int i = 0; i < 8; ++i) {
      int m = row0 + hf * 8 + i, n = ct * 16 + l16;
      float sv = -1.0e30f;
      if (m < L && n < L) sv = acc[i] + Rp[m * L + n];
      Ss[m * 68 + n] = sv;
    }
  }
  __syncthreads();

  // ---- row softmax (49 valid cols), write bf16 P with zero padding ----
  if (tid < 64) {
    int row = tid;
    if (row < L) {
      float mx = -1.0e30f;
      for (int n = 0; n < L; ++n) mx = fmaxf(mx, Ss[row * 68 + n]);
      float sum = 0.0f;
      for (int n = 0; n < L; ++n) {
        float e = __expf(Ss[row * 68 + n] - mx);
        Ss[row * 68 + n] = e;
        sum += e;
      }
      float inv = 1.0f / sum;
      for (int n = 0; n < L; ++n) Ps[(size_t)row * 72 + n] = (__bf16)(Ss[row * 68 + n] * inv);
      for (int n = L; n < 64; ++n) Ps[(size_t)row * 72 + n] = (__bf16)0.0f;
    } else {
      for (int n = 0; n < 64; ++n) Ps[(size_t)row * 72 + n] = (__bf16)0.0f;
    }
  }
  __syncthreads();

  // ---- O = P V ----
  bf16x16 p0 = frag_a(Ps, 72, row0, 0);
  bf16x16 p1 = frag_a(Ps, 72, row0, 32);
#pragma unroll
  for (int ct = 0; ct < 4; ++ct) {
    f32x8 acc = {};
    acc = wmma_bf16(p0, frag_b(Vt, 72, ct * 16, 0), acc);
    acc = wmma_bf16(p1, frag_b(Vt, 72, ct * 16, 32), acc);
#pragma unroll
    for (int i = 0; i < 8; ++i) {
      int m = row0 + hf * 8 + i, d = ct * 16 + l16;
      if (m < L)
        ao[(((size_t)w * L + m) * NHEAD + h) * HD + d] = (__bf16)acc[i];
    }
  }
}

// ---------------------------------------------------------------------------
// Kernel 3: out = ao @ W_out^T + b_out, un-permuted scatter to [B, N, C] f32.
// A panel (bf16) and weights (bf16) staged via async global->LDS.
// ---------------------------------------------------------------------------
__global__ __launch_bounds__(256) void out_proj_kernel(
    const __bf16* __restrict__ ao, const __bf16* __restrict__ wbf,
    const float* __restrict__ b_out, float* __restrict__ out) {
  using namespace cfg;
  __shared__ __bf16 As[32 * 520];
  __shared__ __bf16 Bs[256 * 40];

  const int tid = threadIdx.x;
  const int lane = tid & 31, l16 = lane & 15, hf = lane >> 4;
  const int wv = tid >> 5;
  const int m0 = blockIdx.x * 32;

  // async-stage A panel (already bf16): 128B per thread
  {
    const int arow = tid >> 3;
    const int acol0 = (tid & 7) * 64;
    const __bf16* asrc = ao + (size_t)(m0 + arow) * C;
    const unsigned aOff = lds_off(As) + (unsigned)(arow * 520) * 2;
#pragma unroll
    for (int j = 0; j < 8; ++j) {
      int col = acol0 + j * 8;
      async_g2l_b128(aOff + col * 2, asrc + col);
    }
  }
  wait_async0();
  __syncthreads();

  const unsigned bsOff = lds_off(Bs) + tid * 80;

  for (int nb = 0; nb < C / 256; ++nb) {         // 2 N tiles of 256
    f32x8 acc[2][2] = {};
    const __bf16* bsrc = wbf + (size_t)(nb * 256 + tid) * C;
    for (int kt = 0; kt < C / 32; ++kt) {
#pragma unroll
      for (int j = 0; j < 4; ++j)
        async_g2l_b128(bsOff + j * 16, bsrc + kt * 32 + j * 8);
      wait_async0();
      __syncthreads();
      bf16x16 a0 = frag_a(As, 520, 0, kt * 32);
      bf16x16 a1 = frag_a(As, 520, 16, kt * 32);
      bf16x16 b0 = frag_b(Bs, 40, wv * 32, 0);
      bf16x16 b1 = frag_b(Bs, 40, wv * 32 + 16, 0);
      acc[0][0] = wmma_bf16(a0, b0, acc[0][0]);
      acc[0][1] = wmma_bf16(a0, b1, acc[0][1]);
      acc[1][0] = wmma_bf16(a1, b0, acc[1][0]);
      acc[1][1] = wmma_bf16(a1, b1, acc[1][1]);
      __syncthreads();
    }
#pragma unroll
    for (int mi = 0; mi < 2; ++mi) {
#pragma unroll
      for (int ci = 0; ci < 2; ++ci) {
#pragma unroll
        for (int i = 0; i < 8; ++i) {
          int rr = m0 + mi * 16 + hf * 8 + i;
          int cg = nb * 256 + wv * 32 + ci * 16 + l16;
          float v = acc[mi][ci][i] + b_out[cg];
          int w = rr / L, l = rr % L;
          int b = w / NW, rem = w % NW;
          int qy = rem / QDIM, qx = rem % QDIM;
          int ph = l / WSZ, pw = l % WSZ;
          int n = (qy * WSZ + ph) * IMG + (qx * WSZ + pw);
          out[((size_t)b * (IMG * IMG) + n) * C + cg] = v;
        }
      }
    }
  }
}

// ---------------------------------------------------------------------------
extern "C" void kernel_launch(void* const* d_in, const int* in_sizes, int n_in,
                              void* d_out, int out_size, void* d_ws, size_t ws_size,
                              hipStream_t stream) {
  using namespace cfg;
  (void)in_sizes; (void)n_in; (void)out_size; (void)ws_size;

  const float* x      = (const float*)d_in[0];
  const float* w_in   = (const float*)d_in[1];
  const float* b_in   = (const float*)d_in[2];
  const float* w_out  = (const float*)d_in[3];
  const float* b_out  = (const float*)d_in[4];
  const float* rpe    = (const float*)d_in[5];
  const int*   relidx = (const int*)d_in[6];
  float* out = (float*)d_out;

  __bf16* qkv     = (__bf16*)d_ws;                 // [BW][3][NH][L][HD] bf16
  __bf16* ao      = qkv + (size_t)MTOT * C3;       // [BW][L][NH][HD]    bf16
  __bf16* wbf_in  = ao + (size_t)MTOT * C;         // [1536][512] bf16
  __bf16* wbf_out = wbf_in + (size_t)C3 * C;       // [512][512]  bf16

  convert_weights_kernel<<<dim3((C3 * C / 4 + C * C / 4 + 255) / 256), 256, 0, stream>>>(
      w_in, w_out, wbf_in, wbf_out);
  qkv_proj_kernel<<<dim3(MTOT / 32), 256, 0, stream>>>(x, wbf_in, b_in, qkv);
  attn_kernel<<<dim3(BW * NHEAD), 128, 0, stream>>>(qkv, rpe, relidx, ao);
  out_proj_kernel<<<dim3(MTOT / 32), 256, 0, stream>>>(ao, wbf_out, b_out, out);
}
